// SparseTopKAttention_38122129719509
// MI455X (gfx1250) — matrix operand order repaired and checked by reading, not compiled, not run on hardware
//
#include <hip/hip_runtime.h>
#include <math.h>

// Problem constants (match reference)
#define B_ 2
#define S_ 2048
#define E_ 1024
#define H_ 16
#define K_ 32
#define D_ 64

typedef __attribute__((ext_vector_type(16))) _Float16 v16h;
typedef __attribute__((ext_vector_type(8)))  _Float16 v8h;
typedef __attribute__((ext_vector_type(8)))  float    v8f;
typedef __attribute__((ext_vector_type(4)))  float    v4f;

// ---------------------------------------------------------------------------
// WMMA fragment loaders (CDNA5 16x16x32 f16 layouts, ISA 7.12.2)
// A-matrix 16x32 (MxK): lane L holds row M=L%16.
//   lanes<16: K = {0..7, 16..23}; lanes>=16: K = {8..15, 24..31}
//   -> two 16B chunks at byte offsets 2s and 2s+32 inside the 64B k-window.
__device__ __forceinline__ v16h load_frag_A(const _Float16* __restrict__ row,
                                            int kk, int laneHi) {
  const char* base = (const char*)row + kk * 64 + laneHi * 16;
  v8h lo = *(const v8h*)(base);
  v8h hi = *(const v8h*)(base + 32);
  return __builtin_shufflevector(lo, hi, 0,1,2,3,4,5,6,7,8,9,10,11,12,13,14,15);
}
// B-matrix 32x16 (KxN) in column-major memory (i.e. rows of W / rows of K):
//   lane L holds column N=L%16; lanes<16: K = {0..7, 16..23}; lanes>=16: K = {8..15, 24..31}
//   -> two 16B chunks at byte offsets 16h and 16h+32 inside the 64B k-window (matches A).
__device__ __forceinline__ v16h load_frag_B(const _Float16* __restrict__ row,
                                            int kk, int laneHi) {
  const char* base = (const char*)row + kk * 64 + laneHi * 16;
  v8h lo = *(const v8h*)(base);
  v8h hi = *(const v8h*)(base + 32);
  return __builtin_shufflevector(lo, hi, 0,1,2,3,4,5,6,7,8,9,10,11,12,13,14,15);
}

__device__ __forceinline__ v8f wmma_f16(v16h a, v16h b, v8f c) {
  return __builtin_amdgcn_wmma_f32_16x16x32_f16(false, a, false, b,
                                                (short)0, c, false, false);
}

// ---------------------------------------------------------------------------
// K1: fp32 -> f16 conversion
__global__ void cvt_f32_to_f16(const float* __restrict__ src,
                               _Float16* __restrict__ dst, int n) {
  int i = blockIdx.x * blockDim.x + threadIdx.x;
  if (i < n) dst[i] = (_Float16)src[i];
}

// ---------------------------------------------------------------------------
// K2: fused QKV projection. y = x @ W^T + b, split heads.
// Wh = [Wq|Wk|Wv] f16, each [E,E] row-major (rows are output channel n).
// One wave per 64x64 macro-tile (4x4 WMMA tiles, 16 accumulators):
//   per K-step: 4 A-frags + 4 B-frags loaded, 16 wmma issued (~32 FLOP/B).
// tiles = 3 * (B*S/64) * (E/64) = 3*64*16 = 3072 ; 8 waves/block -> 384 blocks.
__global__ __launch_bounds__(256) void
qkv_gemm(const _Float16* __restrict__ xh, const _Float16* __restrict__ Wh,
         const float* __restrict__ bq, const float* __restrict__ bk,
         const float* __restrict__ bv,
         _Float16* __restrict__ qh, _Float16* __restrict__ kh,
         float* __restrict__ vf) {
  const int lane   = threadIdx.x & 31;
  const int wave   = threadIdx.x >> 5;
  const int tile   = blockIdx.x * 8 + wave;       // < 3072
  const int p      = tile >> 10;                  // 0:q 1:k 2:v
  const int rem    = tile & 1023;
  const int m0     = (rem >> 4) << 6;             // token macro-tile base
  const int n0     = (rem & 15) << 6;             // channel macro-tile base
  const int laneHi = lane >> 4;
  const int col    = lane & 15;

  const _Float16* Abase = xh + (size_t)(m0 + col) * E_;
  const _Float16* Bbase = Wh + (size_t)p * E_ * E_ + (size_t)(n0 + col) * E_;

  v8f acc[16];
#pragma unroll
  for (int u = 0; u < 16; ++u) acc[u] = (v8f){};

#pragma unroll 2
  for (int kk = 0; kk < E_ / 32; ++kk) {
    __builtin_prefetch((const char*)Abase + (kk + 4) * 64, 0, 3);
    __builtin_prefetch((const char*)Bbase + (kk + 4) * 64, 0, 3);
    v16h aF[4], bF[4];
#pragma unroll
    for (int mi = 0; mi < 4; ++mi)
      aF[mi] = load_frag_A(Abase + (size_t)(mi * 16) * E_, kk, laneHi);
#pragma unroll
    for (int ni = 0; ni < 4; ++ni)
      bF[ni] = load_frag_B(Bbase + (size_t)(ni * 16) * E_, kk, laneHi);
#pragma unroll
    for (int mi = 0; mi < 4; ++mi)
#pragma unroll
      for (int ni = 0; ni < 4; ++ni)
        acc[mi * 4 + ni] = wmma_f16(aF[mi], bF[ni], acc[mi * 4 + ni]);
  }

  const float* bias = (p == 0 ? bq : (p == 1 ? bk : bv));
  const int    bb   = m0 / S_;
  const int    sb   = m0 % S_;
  const int    h    = n0 >> 6;                    // 64-wide n-tile = one head

#pragma unroll
  for (int ni = 0; ni < 4; ++ni) {
    const int   n  = n0 + ni * 16 + col;
    const float bn = bias[n];
    const int   d  = n & 63;
    const size_t hd = (((size_t)bb * H_ + h) * S_) * D_ + d;
#pragma unroll
    for (int mi = 0; mi < 4; ++mi) {
#pragma unroll
      for (int i = 0; i < 8; ++i) {
        const int   M = mi * 16 + i + laneHi * 8;
        const float v = acc[mi * 4 + ni][i] + bn;
        const size_t o = hd + (size_t)(sb + M) * D_;
        if      (p == 0) qh[o] = (_Float16)v;
        else if (p == 1) kh[o] = (_Float16)v;
        else             vf[o] = v;
      }
    }
  }
}

// ---------------------------------------------------------------------------
// K3: streaming scores + top-32 + softmax + weighted V.
// One wave per (b*h, 16-query tile). Grid = 32 * 128 = 4096 blocks of 32.
__global__ __launch_bounds__(32) void
attn_topk(const _Float16* __restrict__ qh, const _Float16* __restrict__ kh,
          const float* __restrict__ vf, _Float16* __restrict__ ctxh) {
  __shared__ float stile[16][17];   // 16x16 score tile (+pad)
  __shared__ float topv[16][32];
  __shared__ int   topi[16][32];

  const int lane   = threadIdx.x;
  const int qt     = blockIdx.x & 127;
  const int bh     = blockIdx.x >> 7;             // b*H + h
  const int s0     = qt << 4;
  const int laneHi = lane >> 4;
  const int col    = lane & 15;

  // Q fragment for this wave's 16 query rows, full D=64 (two k-steps).
  const _Float16* qrow = qh + ((size_t)bh * S_ + s0 + col) * D_;
  const v16h a0 = load_frag_A(qrow, 0, laneHi);
  const v16h a1 = load_frag_A(qrow, 1, laneHi);

  if (lane < 16) {
#pragma unroll
    for (int j = 0; j < 32; ++j) { topv[lane][j] = -1.0e30f; topi[lane][j] = 0; }
  }
  float minVal = -1.0e30f;
  int   minPos = 0;
  __syncthreads();

  const float rscale = 0.125f;                    // 1/sqrt(64)
  for (int t = 0; t < S_ / 16; ++t) {
    const _Float16* krow = kh + ((size_t)bh * S_ + t * 16 + col) * D_;
    const v16h b0 = load_frag_B(krow, 0, laneHi);
    const v16h b1 = load_frag_B(krow, 1, laneHi);
    v8f c = {};
    c = wmma_f16(a0, b0, c);
    c = wmma_f16(a1, b1, c);
#pragma unroll
    for (int i = 0; i < 8; ++i) stile[i + laneHi * 8][col] = c[i] * rscale;
    __syncthreads();

    if (lane < 16) {
      const int row = lane;
      for (int j = 0; j < 16; ++j) {
        const float vc = stile[row][j];
        if (vc > minVal) {                         // replace current minimum
          topv[row][minPos] = vc;
          topi[row][minPos] = t * 16 + j;
          minVal = topv[row][0]; minPos = 0;
#pragma unroll
          for (int q = 1; q < 32; ++q) {
            const float tv = topv[row][q];
            if (tv < minVal) { minVal = tv; minPos = q; }
          }
        }
      }
    }
    __syncthreads();
  }

  // softmax over the 32 survivors (per row)
  if (lane < 16) {
    const int row = lane;
    float m = -1.0e30f;
#pragma unroll
    for (int j = 0; j < 32; ++j) m = fmaxf(m, topv[row][j]);
    float sum = 0.f;
#pragma unroll
    for (int j = 0; j < 32; ++j) {
      const float e = __expf(topv[row][j] - m);
      topv[row][j] = e;
      sum += e;
    }
    const float inv = 1.0f / sum;
#pragma unroll
    for (int j = 0; j < 32; ++j) topv[row][j] *= inv;
  }
  __syncthreads();

  // weighted V gather: lane -> (row = lane%16, d-segment = 32*(lane/16))
  const int row  = lane & 15;
  const int dseg = laneHi * 32;
  float acc[32];
#pragma unroll
  for (int u = 0; u < 32; ++u) acc[u] = 0.f;
  const float* vbase = vf + (size_t)bh * S_ * D_ + dseg;
  for (int j = 0; j < 32; ++j) {
    const float p    = topv[row][j];
    const int   kidx = topi[row][j];
    const v4f*  vp   = (const v4f*)(vbase + (size_t)kidx * D_);
#pragma unroll
    for (int u = 0; u < 8; ++u) {
      const v4f vv = vp[u];
      acc[u * 4 + 0] += p * vv[0];
      acc[u * 4 + 1] += p * vv[1];
      acc[u * 4 + 2] += p * vv[2];
      acc[u * 4 + 3] += p * vv[3];
    }
  }
  const int bb = bh / H_;
  const int hh = bh % H_;
  _Float16* op = ctxh + ((size_t)bb * S_ + s0 + row) * E_ + hh * D_ + dseg;
#pragma unroll
  for (int u = 0; u < 32; ++u) op[u] = (_Float16)acc[u];
}

// ---------------------------------------------------------------------------
// K4: output projection out = ctx @ Wo^T + bo (fp32 result to d_out)
// Same 64x64 macro-tile scheme. tiles = 64*16 = 1024 -> 128 blocks.
__global__ __launch_bounds__(256) void
out_gemm(const _Float16* __restrict__ ctxh, const _Float16* __restrict__ Woh,
         const float* __restrict__ bo, float* __restrict__ out) {
  const int lane   = threadIdx.x & 31;
  const int wave   = threadIdx.x >> 5;
  const int tile   = blockIdx.x * 8 + wave;       // < 1024
  const int m0     = (tile >> 4) << 6;
  const int n0     = (tile & 15) << 6;
  const int laneHi = lane >> 4;
  const int col    = lane & 15;

  const _Float16* Abase = ctxh + (size_t)(m0 + col) * E_;
  const _Float16* Bbase = Woh + (size_t)(n0 + col) * E_;

  v8f acc[16];
#pragma unroll
  for (int u = 0; u < 16; ++u) acc[u] = (v8f){};

#pragma unroll 2
  for (int kk = 0; kk < E_ / 32; ++kk) {
    __builtin_prefetch((const char*)Abase + (kk + 4) * 64, 0, 3);
    __builtin_prefetch((const char*)Bbase + (kk + 4) * 64, 0, 3);
    v16h aF[4], bF[4];
#pragma unroll
    for (int mi = 0; mi < 4; ++mi)
      aF[mi] = load_frag_A(Abase + (size_t)(mi * 16) * E_, kk, laneHi);
#pragma unroll
    for (int ni = 0; ni < 4; ++ni)
      bF[ni] = load_frag_B(Bbase + (size_t)(ni * 16) * E_, kk, laneHi);
#pragma unroll
    for (int mi = 0; mi < 4; ++mi)
#pragma unroll
      for (int ni = 0; ni < 4; ++ni)
        acc[mi * 4 + ni] = wmma_f16(aF[mi], bF[ni], acc[mi * 4 + ni]);
  }

#pragma unroll
  for (int ni = 0; ni < 4; ++ni) {
    const int   n  = n0 + ni * 16 + col;
    const float bn = bo[n];
#pragma unroll
    for (int mi = 0; mi < 4; ++mi) {
#pragma unroll
      for (int i = 0; i < 8; ++i) {
        const int M = mi * 16 + i + laneHi * 8;
        out[(size_t)(m0 + M) * E_ + n] = acc[mi * 4 + ni][i] + bn;
      }
    }
  }
}

// ---------------------------------------------------------------------------
extern "C" void kernel_launch(void* const* d_in, const int* in_sizes, int n_in,
                              void* d_out, int out_size, void* d_ws, size_t ws_size,
                              hipStream_t stream) {
  const float* x  = (const float*)d_in[0];
  const float* Wq = (const float*)d_in[1];
  const float* bq = (const float*)d_in[2];
  const float* Wk = (const float*)d_in[3];
  const float* bk = (const float*)d_in[4];
  const float* Wv = (const float*)d_in[5];
  const float* bv = (const float*)d_in[6];
  const float* Wo = (const float*)d_in[7];
  const float* bo = (const float*)d_in[8];

  const size_t ME = (size_t)B_ * S_ * E_;   // 4,194,304
  const size_t EE = (size_t)E_ * E_;        // 1,048,576

  char* w = (char*)d_ws;
  _Float16* xh   = (_Float16*)w;  w += ME * 2;        // x in f16
  _Float16* Wh   = (_Float16*)w;  w += 4 * EE * 2;    // Wq|Wk|Wv|Wo in f16
  _Float16* qh   = (_Float16*)w;  w += ME * 2;        // Q  [B,H,S,D] f16
  _Float16* kh   = (_Float16*)w;  w += ME * 2;        // K  [B,H,S,D] f16
  float*    vf   = (float*)w;     w += ME * 4;        // V  [B,H,S,D] f32
  _Float16* ctxh = (_Float16*)w;  w += ME * 2;        // attn out [B,S,E] f16
  // total ~56 MB

  // K1: conversions
  cvt_f32_to_f16<<<(int)((ME + 255) / 256), 256, 0, stream>>>(x, xh, (int)ME);
  cvt_f32_to_f16<<<(int)((EE + 255) / 256), 256, 0, stream>>>(Wq, Wh + 0 * EE, (int)EE);
  cvt_f32_to_f16<<<(int)((EE + 255) / 256), 256, 0, stream>>>(Wk, Wh + 1 * EE, (int)EE);
  cvt_f32_to_f16<<<(int)((EE + 255) / 256), 256, 0, stream>>>(Wv, Wh + 2 * EE, (int)EE);
  cvt_f32_to_f16<<<(int)((EE + 255) / 256), 256, 0, stream>>>(Wo, Wh + 3 * EE, (int)EE);

  // K2: QKV projections (3072 wave-macro-tiles / 8 waves per block)
  qkv_gemm<<<384, 256, 0, stream>>>(xh, Wh, bq, bk, bv, qh, kh, vf);

  // K3: streaming scores + top-32 + softmax + weighted V
  attn_topk<<<B_ * H_ * (S_ / 16), 32, 0, stream>>>(qh, kh, vf, ctxh);

  // K4: output projection (1024 wave-macro-tiles / 8 per block)
  out_gemm<<<128, 256, 0, stream>>>(ctxh, Wh + 3 * EE, bo, (float*)d_out);
}
